// GNNEncoder2_4389456576913
// MI455X (gfx1250) — matrix operand order repaired and checked by reading, compiled, not run on hardware
//
#include <hip/hip_runtime.h>

// ---------------- problem constants ----------------
enum : int {
  NNODES = 20000,
  NEDGES = 320000,
  EN     = 340000,       // edges + self loops
  KDIM   = 2416,         // C + POS + CTX
  KPAD   = 2432,         // padded to 76*32
  HC     = 512,          // H*C
  CCH    = 128,
  MPAD   = 20032,        // 313 * 64
  OUTD   = 1024,
  NG     = 500,
  POSD   = 240,
  CTXD   = 2048
};

typedef __attribute__((ext_vector_type(16))) __bf16 v16bf;
typedef __attribute__((ext_vector_type(8)))  float  v8f;

// float -> bf16 (round to nearest even), stored as ushort
__device__ __forceinline__ unsigned short f2bf(float f) {
  unsigned u = __float_as_uint(f);
  u += 0x7fffu + ((u >> 16) & 1u);
  return (unsigned short)(u >> 16);
}
// monotonic float<->uint key for atomicMax on floats (handles negatives)
__device__ __forceinline__ unsigned fkey(float f) {
  unsigned u = __float_as_uint(f);
  return (u & 0x80000000u) ? ~u : (u | 0x80000000u);
}
__device__ __forceinline__ float funkey(unsigned k) {
  unsigned u = (k & 0x80000000u) ? (k & 0x7fffffffu) : ~k;
  return __uint_as_float(u);
}

// CDNA5 async global->LDS copy, 16B per lane, tracked by ASYNCcnt.
// GVS addressing: mem = SGPR64 base + signed 32-bit VGPR offset.
__device__ __forceinline__ void async_ld16(unsigned ldsOff, const void* base, unsigned gByteOff) {
  asm volatile("global_load_async_to_lds_b128 %0, %1, %2"
               :: "v"(ldsOff), "v"(gByteOff), "s"(base)
               : "memory");
}
__device__ __forceinline__ void wait_async0() {
  asm volatile("s_wait_asynccnt 0x0" ::: "memory");
}

// ---------------- setup kernels ----------------
__global__ void k_zero(unsigned* p, size_t n) {
  size_t i = (size_t)blockIdx.x * 256 + threadIdx.x;
  if (i < n) p[i] = 0u;
}

// lin_w (3, 2416, 512) f32 -> Wt (3, 512, 2432) bf16, K-padded with zeros
__global__ void k_convw(const float* __restrict__ lin_w, unsigned short* __restrict__ Wt) {
  size_t i = (size_t)blockIdx.x * 256 + threadIdx.x;
  size_t total = (size_t)3 * 512 * KPAD;
  if (i >= total) return;
  int k = (int)(i % KPAD);
  size_t t = i / KPAD;
  int n = (int)(t % 512);
  int l = (int)(t / 512);
  unsigned short v = 0;
  if (k < KDIM) v = f2bf(lin_w[((size_t)l * KDIM + k) * 512 + n]);
  Wt[i] = v;
}

// out_w (128, 1024) f32 -> OWt (1024, 128) bf16
__global__ void k_convow(const float* __restrict__ out_w, unsigned short* __restrict__ OWt) {
  size_t i = (size_t)blockIdx.x * 256 + threadIdx.x;
  if (i >= (size_t)OUTD * CCH) return;
  int k = (int)(i % CCH);
  int n = (int)(i / CCH);
  OWt[i] = f2bf(out_w[(size_t)k * OUTD + n]);
}

// positional encoding -> A cols [128, 368)
__global__ void k_pe(const float* __restrict__ pos, unsigned short* __restrict__ A) {
  size_t i = (size_t)blockIdx.x * 256 + threadIdx.x;
  if (i >= (size_t)NNODES * POSD) return;
  int j = (int)(i % POSD);
  int n = (int)(i / POSD);
  int d  = j / 80;
  int jj = j % 80;
  int fr = (jj < 40) ? jj : jj - 40;
  // div[fr] = exp(-fr * ln(10000)/120)
  float div = __expf(-(float)fr * 0.07675283643313486f);
  float ang = pos[n * 3 + d] * div;
  float v = (jj < 40) ? sinf(ang) : cosf(ang);
  A[(size_t)n * KPAD + CCH + j] = f2bf(v);
}

// context gather -> A cols [368, 2416)
__global__ void k_ctx(const float* __restrict__ ctx, const int* __restrict__ batch,
                      unsigned short* __restrict__ A) {
  size_t i = (size_t)blockIdx.x * 256 + threadIdx.x;
  if (i >= (size_t)NNODES * CTXD) return;
  int k = (int)(i % CTXD);
  int n = (int)(i / CTXD);
  A[(size_t)n * KPAD + CCH + POSD + k] = f2bf(ctx[(size_t)batch[n] * CTXD + k]);
}

// zero K-pad cols for all rows and full M-pad rows
__global__ void k_padA(unsigned short* __restrict__ A) {
  size_t i = (size_t)blockIdx.x * 256 + threadIdx.x;
  size_t p1 = (size_t)MPAD * (KPAD - KDIM);
  if (i < p1) {
    int row = (int)(i / (KPAD - KDIM));
    int col = KDIM + (int)(i % (KPAD - KDIM));
    A[(size_t)row * KPAD + col] = 0;
  } else {
    size_t j = i - p1;
    if (j >= (size_t)(MPAD - NNODES) * KPAD) return;
    int row = NNODES + (int)(j / KPAD);
    int col = (int)(j % KPAD);
    A[(size_t)row * KPAD + col] = 0;
  }
}

// x = W_init[x_types] + b_init
__global__ void k_xinit(const int* __restrict__ xt, const float* __restrict__ Wi,
                        const float* __restrict__ bi, float* __restrict__ X) {
  size_t i = (size_t)blockIdx.x * 256 + threadIdx.x;
  if (i >= (size_t)NNODES * CCH) return;
  int c = (int)(i & (CCH - 1));
  int n = (int)(i >> 7);
  X[i] = Wi[(size_t)xt[n] * CCH + c] + bi[c];
}

// A[:, :128] = bf16(x)
__global__ void k_xbf(const float* __restrict__ X, unsigned short* __restrict__ A) {
  size_t i = (size_t)blockIdx.x * 256 + threadIdx.x;
  if (i >= (size_t)NNODES * CCH) return;
  int c = (int)(i & (CCH - 1));
  int n = (int)(i >> 7);
  A[(size_t)n * KPAD + c] = f2bf(X[i]);
}

// ---------------- main GEMM: H(N,512) = A(N,2432 bf16) @ Wt^T ----------------
// block = 256 thr (8 waves), tile 64(M) x 128(N); wave -> 16x64 (4 WMMAs/step).
// Double-buffered LDS fed by async global->LDS DMA (ASYNCcnt).
__global__ __launch_bounds__(256) void k_gemm(const unsigned short* __restrict__ A,
                                              const unsigned short* __restrict__ Bt, // (512, KPAD) N-major
                                              float* __restrict__ H) {
  __shared__ __attribute__((aligned(16))) unsigned short lA[2][64 * 32];
  __shared__ __attribute__((aligned(16))) unsigned short lB[2][128 * 32]; // col-major: [col*32 + k]
  const int t    = threadIdx.x;
  const int lane = t & 31;
  const int wave = t >> 5;
  const int wm   = wave >> 1;   // 0..3 : 16-row slice
  const int wn   = wave & 1;    // 0..1 : 64-col slice
  const int mBase = blockIdx.x * 64;   // gridDim.x = 313 -> exactly MPAD rows
  const int nBase = blockIdx.y * 128;  // gridDim.y = 4

  // per-thread copy assignment: A 64x32 (1 x 16B), B 128x32 (2 x 16B)
  const int rA = t >> 2, gA = (t & 3) * 8;
  const int cB = t >> 1, gB = (t & 1) * 8;
  const unsigned aByte = (unsigned)(((size_t)(mBase + rA) * KPAD + gA) * 2);
  const unsigned bByte = (unsigned)(((size_t)(nBase + cB) * KPAD + gB) * 2);

  unsigned laOff[2], lbOff0[2], lbOff1[2];
  #pragma unroll
  for (int b = 0; b < 2; ++b) {
    laOff[b]  = (unsigned)(size_t)&lA[b][rA * 32 + gA];
    lbOff0[b] = (unsigned)(size_t)&lB[b][cB * 32 + gB];
    lbOff1[b] = (unsigned)(size_t)&lB[b][cB * 32 + gB + 16];
  }

  v8f acc0 = {}, acc1 = {}, acc2 = {}, acc3 = {};

  const int ar = wm * 16 + (lane & 15);
  const int ak = (lane >> 4) * 16;
  const int bk = (lane >> 4) * 16;
  const int bc = wn * 64 + (lane & 15);

  // prologue: fetch tile 0 into buffer 0
  async_ld16(laOff[0],  A,  aByte);
  async_ld16(lbOff0[0], Bt, bByte);
  async_ld16(lbOff1[0], Bt, bByte + 32);

  const int NSTEP = KPAD / 32;  // 76
  for (int kt = 0; kt < NSTEP; ++kt) {
    wait_async0();        // own async copies for buffer (kt&1) landed
    __syncthreads();      // everyone's copies landed; prior reads of other buffer done
    if (kt + 1 < NSTEP) { // kick DMA for next tile while we compute
      const unsigned kb = (unsigned)((kt + 1) * 32 * 2);
      const int nb = (kt + 1) & 1;
      async_ld16(laOff[nb],  A,  aByte + kb);
      async_ld16(lbOff0[nb], Bt, bByte + kb);
      async_ld16(lbOff1[nb], Bt, bByte + kb + 32);
    }
    const unsigned short* sA = lA[kt & 1];
    const unsigned short* sB = lB[kt & 1];
    union { v16bf v; uint4 q[2]; } a, b0, b1, b2, b3;
    a.q[0]  = *(const uint4*)&sA[ar * 32 + ak];
    a.q[1]  = *(const uint4*)&sA[ar * 32 + ak + 8];
    b0.q[0] = *(const uint4*)&sB[(bc)      * 32 + bk];
    b0.q[1] = *(const uint4*)&sB[(bc)      * 32 + bk + 8];
    b1.q[0] = *(const uint4*)&sB[(bc + 16) * 32 + bk];
    b1.q[1] = *(const uint4*)&sB[(bc + 16) * 32 + bk + 8];
    b2.q[0] = *(const uint4*)&sB[(bc + 32) * 32 + bk];
    b2.q[1] = *(const uint4*)&sB[(bc + 32) * 32 + bk + 8];
    b3.q[0] = *(const uint4*)&sB[(bc + 48) * 32 + bk];
    b3.q[1] = *(const uint4*)&sB[(bc + 48) * 32 + bk + 8];
    acc0 = __builtin_amdgcn_wmma_f32_16x16x32_bf16(false, a.v, false, b0.v, (short)0, acc0, false, false);
    acc1 = __builtin_amdgcn_wmma_f32_16x16x32_bf16(false, a.v, false, b1.v, (short)0, acc1, false, false);
    acc2 = __builtin_amdgcn_wmma_f32_16x16x32_bf16(false, a.v, false, b2.v, (short)0, acc2, false, false);
    acc3 = __builtin_amdgcn_wmma_f32_16x16x32_bf16(false, a.v, false, b3.v, (short)0, acc3, false, false);
  }

  const int rowOff = mBase + wm * 16 + ((lane >> 4) * 8);
  const int col0   = nBase + wn * 64 + (lane & 15);
  #pragma unroll
  for (int i = 0; i < 8; ++i) {
    int r = rowOff + i;
    if (r < NNODES) {
      float* hp = H + (size_t)r * HC + col0;
      hp[0]  = acc0[i];
      hp[16] = acc1[i];
      hp[32] = acc2[i];
      hp[48] = acc3[i];
    }
  }
}

// ---------------- attention coefficients ----------------
// one wave per node: lanes split 4 heads x 8 chunks of 16 channels
__global__ void k_attn(const float* __restrict__ H, const float* __restrict__ sw,
                       const float* __restrict__ dw, float* __restrict__ asrc,
                       float* __restrict__ adst) {
  int wid  = (int)(((size_t)blockIdx.x * 256 + threadIdx.x) >> 5);
  int lane = threadIdx.x & 31;
  if (wid >= NNODES) return;
  int h = lane >> 3, part = lane & 7;
  const float* hp = H + (size_t)wid * HC + h * CCH + part * 16;
  const float* s  = sw + h * CCH + part * 16;
  const float* d  = dw + h * CCH + part * 16;
  float s1 = 0.f, s2 = 0.f;
  #pragma unroll
  for (int j = 0; j < 16; ++j) { float v = hp[j]; s1 += v * s[j]; s2 += v * d[j]; }
  #pragma unroll
  for (int m = 1; m < 8; m <<= 1) { s1 += __shfl_xor(s1, m, 32); s2 += __shfl_xor(s2, m, 32); }
  if (part == 0) { asrc[wid * 4 + h] = s1; adst[wid * 4 + h] = s2; }
}

// ---------------- edge pass A: logit + segment max ----------------
__global__ void k_edge_logit(const int* __restrict__ ei, const float* __restrict__ asrc,
                             const float* __restrict__ adst, float* __restrict__ logits,
                             unsigned* __restrict__ mkey) {
  int e = blockIdx.x * 256 + threadIdx.x;
  if (e >= EN) return;
  int s, d;
  if (e < NEDGES) { s = ei[e]; d = ei[NEDGES + e]; } else { s = d = e - NEDGES; }
  #pragma unroll
  for (int h = 0; h < 4; ++h) {
    float v = asrc[s * 4 + h] + adst[d * 4 + h];
    v = v > 0.f ? v : 0.2f * v;             // leaky_relu(0.2)
    logits[(size_t)e * 4 + h] = v;
    atomicMax(&mkey[d * 4 + h], fkey(v));
  }
}

// ---------------- edge pass B: exp + denom ----------------
__global__ void k_edge_exp(const int* __restrict__ ei, const unsigned* __restrict__ mkey,
                           float* __restrict__ logits, float* __restrict__ denom) {
  int e = blockIdx.x * 256 + threadIdx.x;
  if (e >= EN) return;
  int d;
  if (e < NEDGES) { d = ei[NEDGES + e]; } else { d = e - NEDGES; }
  #pragma unroll
  for (int h = 0; h < 4; ++h) {
    float m  = funkey(mkey[d * 4 + h]);
    float ex = expf(logits[(size_t)e * 4 + h] - m);
    logits[(size_t)e * 4 + h] = ex;
    atomicAdd(&denom[d * 4 + h], ex);
  }
}

// ---------------- edge pass C: normalize + aggregate (head mean fused) ----------------
// one wave per edge; lanes stride channels (coalesced 128B reads of h[src])
__global__ void k_edge_agg(const int* __restrict__ ei, const float* __restrict__ logits,
                           const float* __restrict__ denom, const float* __restrict__ H,
                           float* __restrict__ agg) {
  int wid  = (int)(((size_t)blockIdx.x * 256 + threadIdx.x) >> 5);
  int lane = threadIdx.x & 31;
  if (wid >= EN) return;
  int s, d;
  if (wid < NEDGES) { s = ei[wid]; d = ei[NEDGES + wid]; } else { s = d = wid - NEDGES; }
  float al[4];
  #pragma unroll
  for (int h = 0; h < 4; ++h)
    al[h] = logits[(size_t)wid * 4 + h] / (denom[d * 4 + h] + 1e-16f);
  const float* hp = H + (size_t)s * HC;
  #pragma unroll
  for (int j = 0; j < 4; ++j) {
    int c = lane + 32 * j;
    float v = 0.f;
    #pragma unroll
    for (int h = 0; h < 4; ++h) v += al[h] * hp[h * CCH + c];
    atomicAdd(&agg[(size_t)d * CCH + c], 0.25f * v);   // mean over 4 heads
  }
}

// ---------------- BN reduce / finalize / update ----------------
__global__ void k_bnred(const float* __restrict__ agg, const float* __restrict__ bias,
                        float* __restrict__ stats) {
  __shared__ float ssum[256], ssq[256];
  int t = threadIdx.x;
  int c = t & 127;
  float s = 0.f, q = 0.f;
  for (int r = blockIdx.x * 2 + (t >> 7); r < NNODES; r += gridDim.x * 2) {
    float v = agg[(size_t)r * CCH + c] + bias[c];
    s += v; q += v * v;
  }
  ssum[t] = s; ssq[t] = q;
  __syncthreads();
  if (t < 128) {
    atomicAdd(&stats[c],       ssum[t] + ssum[t + 128]);
    atomicAdd(&stats[128 + c], ssq[t]  + ssq[t + 128]);
  }
}

__global__ void k_bnfin(const float* __restrict__ stats, float* __restrict__ musig) {
  int c = threadIdx.x;  // 128 threads
  float mu  = stats[c] * (1.f / NNODES);
  float var = stats[128 + c] * (1.f / NNODES) - mu * mu;
  musig[c]       = mu;
  musig[128 + c] = rsqrtf(var + 1e-5f);
}

__global__ void k_update(const float* __restrict__ agg, const float* __restrict__ bias,
                         const float* __restrict__ musig, const float* __restrict__ gamma,
                         const float* __restrict__ beta, float* __restrict__ X) {
  size_t i = (size_t)blockIdx.x * 256 + threadIdx.x;
  if (i >= (size_t)NNODES * CCH) return;
  int c = (int)(i & (CCH - 1));
  float v  = agg[i] + bias[c];
  float xn = (v - musig[c]) * musig[128 + c] * gamma[c] + beta[c];
  float g  = 0.5f * xn * (1.f + erff(xn * 0.70710678118654752f)); // exact gelu
  X[i] += g;
}

// ---------------- final GEMM + bias + graph segment-sum ----------------
// wave per 16x16 tile; A = bf16 x (lda=KPAD, first 128 cols), B = OWt (1024,128)
__global__ __launch_bounds__(256) void k_final(const unsigned short* __restrict__ A,
                                               const unsigned short* __restrict__ OWt,
                                               const float* __restrict__ out_b,
                                               const int* __restrict__ batch,
                                               float* __restrict__ out) {
  int wid  = (int)(((size_t)blockIdx.x * 256 + threadIdx.x) >> 5);
  int lane = threadIdx.x & 31;
  const int NT = OUTD / 16;  // 64
  int mt = wid / NT, nt = wid % NT;
  if (mt >= NNODES / 16) return;  // 1250 tiles, 20000 % 16 == 0
  int mBase = mt * 16, nBase = nt * 16;
  const int ar = lane & 15, ak = (lane >> 4) * 16;
  const int bc = nBase + (lane & 15), bk = (lane >> 4) * 16;
  v8f acc = {};
  #pragma unroll
  for (int k0 = 0; k0 < CCH; k0 += 32) {
    union { v16bf v; uint4 q[2]; } a, b;
    const unsigned short* ap = A + (size_t)(mBase + ar) * KPAD + k0;
    a.q[0] = *(const uint4*)(ap + ak);
    a.q[1] = *(const uint4*)(ap + ak + 8);
    const unsigned short* bp = OWt + (size_t)bc * CCH + k0 + bk;
    b.q[0] = *(const uint4*)(bp);
    b.q[1] = *(const uint4*)(bp + 8);
    acc = __builtin_amdgcn_wmma_f32_16x16x32_bf16(false, a.v, false, b.v, (short)0, acc, false, false);
  }
  int row0 = mBase + ((lane >> 4) * 8);
  int col  = nBase + (lane & 15);
  float ob = out_b[col];
  #pragma unroll
  for (int i = 0; i < 8; ++i) {
    int r = row0 + i;
    int g = batch[r];
    atomicAdd(&out[(size_t)g * OUTD + col], acc[i] + ob);
  }
}

// ---------------- host launch ----------------
extern "C" void kernel_launch(void* const* d_in, const int* in_sizes, int n_in,
                              void* d_out, int out_size, void* d_ws, size_t ws_size,
                              hipStream_t stream) {
  const int*   x_types  = (const int*)  d_in[0];
  const float* pos      = (const float*)d_in[1];
  const int*   ei       = (const int*)  d_in[2];
  /* d_in[3] edge_attr: unused by the reference */
  const int*   batch    = (const int*)  d_in[4];
  const float* ctx      = (const float*)d_in[5];
  const float* W_init   = (const float*)d_in[6];
  const float* b_init   = (const float*)d_in[7];
  const float* lin_w    = (const float*)d_in[8];
  const float* att_src  = (const float*)d_in[9];
  const float* att_dst  = (const float*)d_in[10];
  const float* gat_bias = (const float*)d_in[11];
  const float* bn_gamma = (const float*)d_in[12];
  const float* bn_beta  = (const float*)d_in[13];
  const float* out_w    = (const float*)d_in[14];
  const float* out_b    = (const float*)d_in[15];
  float* out = (float*)d_out;

  char* ws = (char*)d_ws;
  size_t off = 0;
  auto carve = [&](size_t bytes) -> char* {
    char* p = ws + off;
    off = (off + bytes + 255) & ~(size_t)255;
    return p;
  };
  unsigned short* A   = (unsigned short*)carve((size_t)MPAD * KPAD * 2);   // 97.4 MB
  unsigned short* Wt  = (unsigned short*)carve((size_t)3 * 512 * KPAD * 2);
  unsigned short* OWt = (unsigned short*)carve((size_t)OUTD * CCH * 2);
  float* H    = (float*)carve((size_t)NNODES * HC * 4);                    // 41 MB
  float* X    = (float*)carve((size_t)NNODES * CCH * 4);
  float* ASRC = (float*)carve((size_t)NNODES * 4 * 4);
  float* ADST = (float*)carve((size_t)NNODES * 4 * 4);
  float* LOG  = (float*)carve((size_t)EN * 4 * 4);
  // contiguous zero region: maxkey | denom | agg | stats
  const size_t ZB_MK = (size_t)NNODES * 4 * 4;         // 320000
  const size_t ZB_DN = (size_t)NNODES * 4 * 4;         // 320000
  const size_t ZB_AG = (size_t)NNODES * CCH * 4;       // 10240000
  const size_t ZB_ST = 1024;
  char* ZR = carve(ZB_MK + ZB_DN + ZB_AG + ZB_ST);
  unsigned* MK  = (unsigned*)ZR;
  float*    DEN = (float*)(ZR + ZB_MK);
  float*    AGG = (float*)(ZR + ZB_MK + ZB_DN);
  float*    ST  = (float*)(ZR + ZB_MK + ZB_DN + ZB_AG);
  float*    MS  = (float*)carve(1024);
  const size_t ZR_U32 = (ZB_MK + ZB_DN + ZB_AG + ZB_ST) / 4;

  auto blocks = [](size_t n) { return (unsigned)((n + 255) / 256); };

  // one-time setup (per call)
  k_convw <<<blocks((size_t)3 * 512 * KPAD), 256, 0, stream>>>(lin_w, Wt);
  k_convow<<<blocks((size_t)OUTD * CCH),     256, 0, stream>>>(out_w, OWt);
  k_pe    <<<blocks((size_t)NNODES * POSD),  256, 0, stream>>>(pos, A);
  k_ctx   <<<blocks((size_t)NNODES * CTXD),  256, 0, stream>>>(ctx, batch, A);
  k_padA  <<<blocks((size_t)MPAD * (KPAD - KDIM) + (size_t)(MPAD - NNODES) * KPAD), 256, 0, stream>>>(A);
  k_xinit <<<blocks((size_t)NNODES * CCH),   256, 0, stream>>>(x_types, W_init, b_init, X);

  for (int l = 0; l < 3; ++l) {
    k_xbf <<<blocks((size_t)NNODES * CCH), 256, 0, stream>>>(X, A);
    k_zero<<<blocks(ZR_U32), 256, 0, stream>>>((unsigned*)ZR, ZR_U32);
    k_gemm<<<dim3(313, 4), 256, 0, stream>>>(A, Wt + (size_t)l * 512 * KPAD, H);
    k_attn<<<blocks((size_t)NNODES * 32), 256, 0, stream>>>(H, att_src + l * HC, att_dst + l * HC, ASRC, ADST);
    k_edge_logit<<<blocks(EN), 256, 0, stream>>>(ei, ASRC, ADST, LOG, MK);
    k_edge_exp  <<<blocks(EN), 256, 0, stream>>>(ei, MK, LOG, DEN);
    k_edge_agg  <<<blocks((size_t)EN * 32), 256, 0, stream>>>(ei, LOG, DEN, H, AGG);
    k_bnred <<<160, 256, 0, stream>>>(AGG, gat_bias + l * CCH, ST);
    k_bnfin <<<1, 128, 0, stream>>>(ST, MS);
    k_update<<<blocks((size_t)NNODES * CCH), 256, 0, stream>>>(AGG, gat_bias + l * CCH, MS,
                                                               bn_gamma + l * CCH, bn_beta + l * CCH, X);
  }

  // final projection + per-graph segment sum
  k_xbf <<<blocks((size_t)NNODES * CCH), 256, 0, stream>>>(X, A);
  k_zero<<<blocks((size_t)NG * OUTD), 256, 0, stream>>>((unsigned*)out, (size_t)NG * OUTD);
  k_final<<<blocks((size_t)(NNODES / 16) * (OUTD / 16) * 32), 256, 0, stream>>>(A, OWt, out_b, batch, out);
}
